// FlowProcess_84009560310173
// MI455X (gfx1250) — matrix operand order repaired and checked by reading, not compiled, not run on hardware
//
#include <hip/hip_runtime.h>
#include <hip/hip_bf16.h>
#include <math.h>

// ---------------------------------------------------------------------------
// MI455X (gfx1250) fused implementation.
//   K1 prep_frags : fp32 weights -> bf16 hi/lo WMMA A-fragments (L2 resident)
//   K2 front      : dwconv3x3 + two 1x1 convs (bf16x3 split WMMA) + 4x4 pool
//                   + relu/sigmoid  ->  h, att_d  [4,256,64,64]
//                   x halo tiles staged with GLOBAL_LOAD_ASYNC_TO_LDS_B32
//   K3 sample_mul : bilinear grid_sample * att_d (L2-resident gather)
//   K4 iter_gemm  : agg' = W1@agg + W2@cur  (M=256,K=512,N=16384, bf16x3 WMMA)
// ---------------------------------------------------------------------------

typedef __attribute__((ext_vector_type(16))) __bf16 v16bf;
typedef __attribute__((ext_vector_type(8)))  float  v8f;

#define CH        256
#define HW_FULL   65536          // 256*256
#define HW_POOL   4096           // 64*64
#define SZ_POOL   (4u * CH * HW_POOL)
#define FRAG_U32  32768u         // per (matrix, hi|lo): 8 kt * 16 mt * 32 lane * 8 dw

union FragBits { uint4 q[2]; unsigned u[8]; v16bf v; };

__device__ inline unsigned short bf_hi(float f) {
  unsigned u = __builtin_bit_cast(unsigned, f);
  u = u + 0x7fffu + ((u >> 16) & 1u);          // round-to-nearest-even to bf16
  return (unsigned short)(u >> 16);
}
__device__ inline float bf2f(unsigned short h) {
  return __builtin_bit_cast(float, ((unsigned)h) << 16);
}
__device__ inline void split_bf(float f, unsigned short& h, unsigned short& l) {
  h = bf_hi(f);
  l = bf_hi(f - bf2f(h));                      // residual also exactly a bf16
}
__device__ inline v16bf frag_from(const unsigned* p) {
  FragBits b;
  b.q[0] = ((const uint4*)p)[0];
  b.q[1] = ((const uint4*)p)[1];
  return b.v;
}
__device__ inline v8f wmma_bf16(v16bf a, v16bf b, v8f c) {
  return __builtin_amdgcn_wmma_f32_16x16x32_bf16(false, a, false, b,
                                                 (short)0, c, false, false);
}
// fp32-ish product via bf16 split: D += Ah*Bh + Ah*Bl + Al*Bh
__device__ inline v8f mm3(v16bf ah, v16bf al, v16bf bh, v16bf bl, v8f c) {
  c = wmma_bf16(ah, bh, c);
  c = wmma_bf16(ah, bl, c);
  c = wmma_bf16(al, bh, c);
  return c;
}

// ---------------------------------------------------------------------------
// K1: weights -> A-fragment layout.
// A (16-bit, 16x32): lane L, vgpr v, elem e:
//   M = mt*16 + (L&15)
//   K = kt*32 + (v<4?0:16) + (L>=16?8:0) + (v&3)*2 + e
// mats: 0=conv_w[256x256] 1=att_pw[256x256] 2=add_w[:, :256] 3=add_w[:,256:]
// ---------------------------------------------------------------------------
__global__ __launch_bounds__(256)
void prep_frags(const float* __restrict__ cw, const float* __restrict__ apw,
                const float* __restrict__ aw, unsigned* __restrict__ frag)
{
  int t    = blockIdx.x * 256 + threadIdx.x;   // 4*8*16*32 = 16384 threads
  int lane = t & 31;
  int mt   = (t >> 5) & 15;
  int kt   = (t >> 9) & 7;
  int mat  = t >> 12;

  const float* W; int ldk, koff;
  if      (mat == 0) { W = cw;  ldk = 256; koff = 0;   }
  else if (mat == 1) { W = apw; ldk = 256; koff = 0;   }
  else if (mat == 2) { W = aw;  ldk = 512; koff = 0;   }
  else               { W = aw;  ldk = 512; koff = 256; }

  int m = mt * 16 + (lane & 15);
  unsigned* ph = frag + (size_t)(mat * 2 + 0) * FRAG_U32 + ((kt * 16 + mt) * 32 + lane) * 8;
  unsigned* pl = frag + (size_t)(mat * 2 + 1) * FRAG_U32 + ((kt * 16 + mt) * 32 + lane) * 8;

  #pragma unroll
  for (int v = 0; v < 8; ++v) {
    unsigned hp = 0, lp = 0;
    #pragma unroll
    for (int e = 0; e < 2; ++e) {
      int k = kt * 32 + ((v < 4) ? 0 : 16) + ((lane >= 16) ? 8 : 0) + (v & 3) * 2 + e;
      float f = W[(size_t)m * ldk + koff + k];
      unsigned short hh, ll; split_bf(f, hh, ll);
      hp |= (unsigned)hh << (e * 16);
      lp |= (unsigned)ll << (e * 16);
    }
    ph[v] = hp; pl[v] = lp;
  }
}

// ---------------------------------------------------------------------------
// K2: fused front end. Block = 256 thr (8 waves): one 8x8 spatial tile
// (= 2x2 pool blocks = 4 WMMA N-tiles) x 128 output channels, both GEMMs.
// B (16-bit, 32x16): lane L, vgpr v, elem e:
//   N = L&15 ; K = (v<4?0:16) + (L>=16?8:0) + (v&3)*2 + e
// Each N-tile is one 4x4 pool block -> pool = shfl_xor max over 16-lane halves.
// ---------------------------------------------------------------------------
__global__ __launch_bounds__(256)
void front_kernel(const float* __restrict__ x, const float* __restrict__ dw,
                  const unsigned* __restrict__ cwh, const unsigned* __restrict__ cwl,
                  const unsigned* __restrict__ pwh, const unsigned* __restrict__ pwl,
                  float* __restrict__ hOut, float* __restrict__ attOut)
{
  __shared__ float xs[32][10][10];                                  // halo tile
  __shared__ __align__(16) unsigned b1h[4][32][8], b1l[4][32][8];   // x  frags
  __shared__ __align__(16) unsigned b2h[4][32][8], b2l[4][32][8];   // dw frags

  const int tid  = threadIdx.x;
  const int lane = tid & 31;
  const int wv   = tid >> 5;

  int bid = blockIdx.x;                    // 4 n * 32 ty * 32 tx * 2 mhalf
  const int mhalf = bid & 1;  bid >>= 1;
  const int tx = bid & 31;
  const int ty = (bid >> 5) & 31;
  const int n  = bid >> 10;

  const float* xg    = x + (size_t)n * CH * HW_FULL;
  const size_t obase = (size_t)n * CH * HW_POOL;

  // fragment-fill decomposition: 2 threads per (nt, lane) slot
  const int f_nt   = tid >> 6;
  const int f_lane = (tid >> 1) & 31;
  const int f_half = tid & 1;
  const int f_p    = f_lane & 15;
  const int f_py   = ((f_nt >> 1) << 2) + (f_p >> 2);
  const int f_px   = ((f_nt & 1) << 2) + (f_p & 3);
  const int f_kb   = (f_lane >= 16) ? 8 : 0;

  const int mtG = mhalf * 8 + wv;

  v8f zero = {0.f,0.f,0.f,0.f,0.f,0.f,0.f,0.f};
  v8f acc1[4], acc2[4];
  #pragma unroll
  for (int i = 0; i < 4; ++i) { acc1[i] = zero; acc2[i] = zero; }

  for (int kc8 = 0; kc8 < 8; ++kc8) {        // K in chunks of 32 channels
    const int kc = kc8 * 32;
    __syncthreads();
    // stage 10x10 x32ch halo tile: async global->LDS copy (no VGPR round trip,
    // tracked by ASYNCcnt); zero-pad border lanes with plain ds stores.
    for (int i = tid; i < 3200; i += 256) {
      int cc = i / 100, r = i % 100, yy = r / 10, xx = r % 10;
      int gy = ty * 8 - 1 + yy, gx = tx * 8 - 1 + xx;
      unsigned lds = (unsigned)(size_t)&xs[cc][yy][xx];
      if ((unsigned)gy < 256u && (unsigned)gx < 256u) {
        unsigned long long ga = (unsigned long long)(size_t)
            (xg + (size_t)(kc + cc) * HW_FULL + gy * 256 + gx);
        asm volatile("global_load_async_to_lds_b32 %0, %1, off"
                     :: "v"(lds), "v"(ga) : "memory");
      } else {
        xs[cc][yy][xx] = 0.f;
      }
    }
    asm volatile("s_wait_asynccnt 0x0" ::: "memory");
    __syncthreads();
    // build B fragments: x (GEMM1) and depthwise-3x3(x) (GEMM2), hi/lo bf16
    for (int v = f_half * 4; v < f_half * 4 + 4; ++v) {
      unsigned p1h = 0, p1l = 0, p2h = 0, p2l = 0;
      #pragma unroll
      for (int e = 0; e < 2; ++e) {
        int c = f_kb + ((v < 4) ? 0 : 16) + (v & 3) * 2 + e;
        float xv = xs[c][f_py + 1][f_px + 1];
        unsigned short hh, ll; split_bf(xv, hh, ll);
        const float* dwp = dw + (size_t)(kc + c) * 9;
        float zv = 0.f;
        #pragma unroll
        for (int ky = 0; ky < 3; ++ky)
          #pragma unroll
          for (int kx = 0; kx < 3; ++kx)
            zv += xs[c][f_py + ky][f_px + kx] * dwp[ky * 3 + kx];
        unsigned short zh, zl; split_bf(zv, zh, zl);
        p1h |= (unsigned)hh << (e * 16); p1l |= (unsigned)ll << (e * 16);
        p2h |= (unsigned)zh << (e * 16); p2l |= (unsigned)zl << (e * 16);
      }
      b1h[f_nt][f_lane][v] = p1h; b1l[f_nt][f_lane][v] = p1l;
      b2h[f_nt][f_lane][v] = p2h; b2l[f_nt][f_lane][v] = p2l;
    }
    __syncthreads();
    // A fragments straight from L2-resident preprocessed weights
    const int abase = ((kc8 * 16 + mtG) * 32 + lane) * 8;
    v16bf a1h = frag_from(cwh + abase), a1l = frag_from(cwl + abase);
    v16bf a2h = frag_from(pwh + abase), a2l = frag_from(pwl + abase);
    #pragma unroll
    for (int nt = 0; nt < 4; ++nt) {
      v16bf B1h = frag_from(&b1h[nt][lane][0]);
      v16bf B1l = frag_from(&b1l[nt][lane][0]);
      acc1[nt] = mm3(a1h, a1l, B1h, B1l, acc1[nt]);
      v16bf B2h = frag_from(&b2h[nt][lane][0]);
      v16bf B2l = frag_from(&b2l[nt][lane][0]);
      acc2[nt] = mm3(a2h, a2l, B2h, B2l, acc2[nt]);
    }
  }

  // epilogue: max over the 16 pixels of each pool block (lane halves), then
  // relu(maxpool(y1)) and sigmoid->maxpool == maxpool->sigmoid for y2.
  #pragma unroll
  for (int nt = 0; nt < 4; ++nt) {
    const int py2 = ty * 2 + (nt >> 1);
    const int px2 = tx * 2 + (nt & 1);
    #pragma unroll
    for (int r = 0; r < 8; ++r) {
      float v1 = acc1[nt][r], v2 = acc2[nt][r];
      v1 = fmaxf(v1, __shfl_xor(v1, 1, 32));  v2 = fmaxf(v2, __shfl_xor(v2, 1, 32));
      v1 = fmaxf(v1, __shfl_xor(v1, 2, 32));  v2 = fmaxf(v2, __shfl_xor(v2, 2, 32));
      v1 = fmaxf(v1, __shfl_xor(v1, 4, 32));  v2 = fmaxf(v2, __shfl_xor(v2, 4, 32));
      v1 = fmaxf(v1, __shfl_xor(v1, 8, 32));  v2 = fmaxf(v2, __shfl_xor(v2, 8, 32));
      if ((lane & 15) == 0) {
        int m = mhalf * 128 + wv * 16 + r + ((lane >= 16) ? 8 : 0);
        size_t o = obase + (size_t)m * HW_POOL + py2 * 64 + px2;
        hOut[o]   = fmaxf(v1, 0.f);
        attOut[o] = 1.f / (1.f + __expf(-v2));
      }
    }
  }
}

// ---------------------------------------------------------------------------
// K3: cur' = grid_sample(cur) * att_d   (bilinear, align_corners, zero pad)
// One thread = one pooled pixel x 32 channels; all data is L2-resident.
// ---------------------------------------------------------------------------
__global__ __launch_bounds__(256)
void sample_mul(const float* __restrict__ cur, const float* __restrict__ att,
                const float* __restrict__ grd, float* __restrict__ out)
{
  int t  = blockIdx.x * 256 + threadIdx.x;   // 16384 px * 8 chunks
  int p  = t >> 3;
  int c0 = (t & 7) * 32;
  int n  = p >> 12;
  int pix = p & 4095;

  const float* g = grd + (size_t)p * 2;
  float ix = (g[0] + 1.f) * 0.5f * 63.f;
  float iy = (g[1] + 1.f) * 0.5f * 63.f;
  float fx0 = floorf(ix), fy0 = floorf(iy);
  int x0 = (int)fx0, y0 = (int)fy0, x1 = x0 + 1, y1 = y0 + 1;
  float wx1 = ix - fx0, wx0 = 1.f - wx1;
  float wy1 = iy - fy0, wy0 = 1.f - wy1;
  float vx0 = (x0 >= 0 && x0 < 64) ? 1.f : 0.f;
  float vx1 = (x1 >= 0 && x1 < 64) ? 1.f : 0.f;
  float vy0 = (y0 >= 0 && y0 < 64) ? 1.f : 0.f;
  float vy1 = (y1 >= 0 && y1 < 64) ? 1.f : 0.f;
  int xc0 = min(max(x0, 0), 63), xc1 = min(max(x1, 0), 63);
  int yc0 = min(max(y0, 0), 63), yc1 = min(max(y1, 0), 63);
  float w00 = wy0 * wx0 * vy0 * vx0, w01 = wy0 * wx1 * vy0 * vx1;
  float w10 = wy1 * wx0 * vy1 * vx0, w11 = wy1 * wx1 * vy1 * vx1;
  int o00 = yc0 * 64 + xc0, o01 = yc0 * 64 + xc1;
  int o10 = yc1 * 64 + xc0, o11 = yc1 * 64 + xc1;

  size_t nb = (size_t)n * CH * HW_POOL;
  #pragma unroll 4
  for (int c = c0; c < c0 + 32; ++c) {
    const float* cp = cur + nb + (size_t)c * HW_POOL;
    float v = w00 * cp[o00] + w01 * cp[o01] + w10 * cp[o10] + w11 * cp[o11];
    size_t oi = nb + (size_t)c * HW_POOL + pix;
    out[oi] = v * att[oi];
  }
}

// ---------------------------------------------------------------------------
// K4: agg' = W1 @ agg + W2 @ cur'  (M=256, K=512, N=16384)
// Block = 256 thr, 64 consecutive pixels; wave wv owns M-tiles {wv, wv+8}.
// ---------------------------------------------------------------------------
__global__ __launch_bounds__(256)
void iter_gemm(const float* __restrict__ aggIn, const float* __restrict__ curIn,
               const unsigned* __restrict__ w1h, const unsigned* __restrict__ w1l,
               const unsigned* __restrict__ w2h, const unsigned* __restrict__ w2l,
               float* __restrict__ aggOut)
{
  __shared__ __align__(16) unsigned bh[4][32][8], bl[4][32][8];

  const int tid  = threadIdx.x;
  const int lane = tid & 31;
  const int wv   = tid >> 5;

  const int pBase = blockIdx.x * 64;
  const int n     = pBase >> 12;
  const size_t nbase = (size_t)n * CH * HW_POOL;
  const int pixIn = pBase & 4095;

  const int f_nt   = tid >> 6;
  const int f_lane = (tid >> 1) & 31;
  const int f_half = tid & 1;
  const int f_pp   = pixIn + f_nt * 16 + (f_lane & 15);
  const int f_kb   = (f_lane >= 16) ? 8 : 0;

  v8f zero = {0.f,0.f,0.f,0.f,0.f,0.f,0.f,0.f};
  v8f acc[2][4];
  #pragma unroll
  for (int a = 0; a < 2; ++a)
    #pragma unroll
    for (int b = 0; b < 4; ++b) acc[a][b] = zero;

  for (int kt = 0; kt < 16; ++kt) {          // K=512 in chunks of 32
    __syncthreads();
    const float* S = (kt < 8) ? aggIn : curIn;   // concat([agg, cur]) on K
    const int cbase = (kt & 7) * 32;
    for (int v = f_half * 4; v < f_half * 4 + 4; ++v) {
      unsigned ph = 0, pl = 0;
      #pragma unroll
      for (int e = 0; e < 2; ++e) {
        int c = cbase + f_kb + ((v < 4) ? 0 : 16) + (v & 3) * 2 + e;
        float f = S[nbase + (size_t)c * HW_POOL + f_pp];
        unsigned short hh, ll; split_bf(f, hh, ll);
        ph |= (unsigned)hh << (e * 16);
        pl |= (unsigned)ll << (e * 16);
      }
      bh[f_nt][f_lane][v] = ph;
      bl[f_nt][f_lane][v] = pl;
    }
    __syncthreads();

    const unsigned* WH = (kt < 8) ? w1h : w2h;
    const unsigned* WL = (kt < 8) ? w1l : w2l;
    const int ktm = kt & 7;
    v16bf aH[2], aL[2];
    #pragma unroll
    for (int m2 = 0; m2 < 2; ++m2) {
      const int abase = ((ktm * 16 + (wv + m2 * 8)) * 32 + lane) * 8;
      aH[m2] = frag_from(WH + abase);
      aL[m2] = frag_from(WL + abase);
    }
    #pragma unroll
    for (int nt = 0; nt < 4; ++nt) {
      v16bf Bh = frag_from(&bh[nt][lane][0]);
      v16bf Bl = frag_from(&bl[nt][lane][0]);
      #pragma unroll
      for (int m2 = 0; m2 < 2; ++m2)
        acc[m2][nt] = mm3(aH[m2], aL[m2], Bh, Bl, acc[m2][nt]);
    }
  }

  #pragma unroll
  for (int m2 = 0; m2 < 2; ++m2) {
    const int cBase = (wv + m2 * 8) * 16 + ((lane >= 16) ? 8 : 0);
    #pragma unroll
    for (int nt = 0; nt < 4; ++nt) {
      const int pp = pixIn + nt * 16 + (lane & 15);
      #pragma unroll
      for (int r = 0; r < 8; ++r)
        aggOut[nbase + (size_t)(cBase + r) * HW_POOL + pp] = acc[m2][nt][r];
    }
  }
}

// ---------------------------------------------------------------------------
extern "C" void kernel_launch(void* const* d_in, const int* in_sizes, int n_in,
                              void* d_out, int out_size, void* d_ws, size_t ws_size,
                              hipStream_t stream)
{
  const float* x   = (const float*)d_in[0];   // [4,256,256,256]
  const float* grd = (const float*)d_in[1];   // [4,64,64,2]
  const float* adw = (const float*)d_in[2];   // [256,1,3,3]
  const float* apw = (const float*)d_in[3];   // [256,256,1,1]
  const float* cw  = (const float*)d_in[4];   // [256,256,1,1]
  const float* aw  = (const float*)d_in[5];   // [256,512,1,1]
  float* outp = (float*)d_out;                // [4,256,64,64]

  // workspace: 8 frag banks (1 MB) + 5 pooled-size float buffers (~84 MB)
  unsigned* frag = (unsigned*)d_ws;
  float* fbuf = (float*)(frag + 8u * FRAG_U32);
  float* hBuf = fbuf;
  float* attB = fbuf + (size_t)1 * SZ_POOL;
  float* curA = fbuf + (size_t)2 * SZ_POOL;
  float* curB = fbuf + (size_t)3 * SZ_POOL;
  float* aggB = fbuf + (size_t)4 * SZ_POOL;

  const unsigned* cwh = frag + 0 * FRAG_U32;
  const unsigned* cwl = frag + 1 * FRAG_U32;
  const unsigned* pwh = frag + 2 * FRAG_U32;
  const unsigned* pwl = frag + 3 * FRAG_U32;
  const unsigned* w1h = frag + 4 * FRAG_U32;
  const unsigned* w1l = frag + 5 * FRAG_U32;
  const unsigned* w2h = frag + 6 * FRAG_U32;
  const unsigned* w2l = frag + 7 * FRAG_U32;

  prep_frags  <<<64,   256, 0, stream>>>(cw, apw, aw, frag);
  front_kernel<<<8192, 256, 0, stream>>>(x, adw, cwh, cwl, pwh, pwl, hBuf, attB);

  const float* aggIn = hBuf;   // aggregate = h
  const float* curIn = hBuf;   // cur       = h
  for (int i = 0; i < 10; ++i) {
    float* curOut = (i & 1) ? curB : curA;
    sample_mul<<<512, 256, 0, stream>>>(curIn, attB, grd, curOut);
    float* aggOut = (i == 9) ? outp : ((i & 1) ? hBuf : aggB);  // hBuf dead after i=0
    iter_gemm<<<256, 256, 0, stream>>>(aggIn, curOut, w1h, w1l, w2h, w2l, aggOut);
    aggIn = aggOut;
    curIn = curOut;
  }
}
